// SimulatorModel_34179349741862
// MI455X (gfx1250) — matrix operand order repaired and checked by reading, not compiled, not run on hardware
//
#include <hip/hip_runtime.h>
#include <hip/hip_bf16.h>

// GNN simulator on gfx1250. All small MLPs mapped to v_wmma_f32_16x16x32_f16:
// M = tile of 16 edges/nodes, K padded to 32/64, N padded to 16.
// Weights prepacked into WMMA-B fragment order; biases folded into a spare
// K row (feature value 1.0) so first-GEMM accumulators start at literal 0.
// A fragments assembled in LDS (row-major f16) and loaded as 2x ds_load_b128.

typedef __attribute__((ext_vector_type(16))) _Float16 v16h;
typedef __attribute__((ext_vector_type(8)))  float    v8f;
typedef __attribute__((ext_vector_type(4)))  unsigned int v4u;

union V16U { v4u u[2]; v16h h; };

#define WV 8   // waves per block (256 threads, wave32)

__device__ __forceinline__ v8f wmma_f16(v16h a, v16h b, v8f c) {
  // D = A(16x32 f16) * B(32x16 f16) + C(16x16 f32)
  return __builtin_amdgcn_wmma_f32_16x16x32_f16(false, a, false, b, (short)0, c,
                                                false, false);
}

// A-fragment (16x32 f16) from row-major LDS tile [16][rowHalves].
// Lane l: m = l&15, khalf = l>>4; elements 0..7  <- K = kOff + khalf*8 + 0..7
//                                 elements 8..15 <- K = kOff + 16 + khalf*8 + 0..7
// Both runs are 16B contiguous -> two ds_load_b128.
__device__ __forceinline__ v16h lds_a_frag(const _Float16* base, int rowHalves,
                                           int kOff, int lane) {
  int m = lane & 15, kh = lane >> 4;
  const _Float16* r = base + m * rowHalves + kOff + kh * 8;
  V16U v;
  v.u[0] = *(const v4u*)(r);
  v.u[1] = *(const v4u*)(r + 16);
  return v.h;
}

// Prepacked B fragment: 32B per lane, contiguous global load.
__device__ __forceinline__ v16h ld_b_frag(const _Float16* p, int lane) {
  return ((const v16h*)p)[lane];
}

__device__ __forceinline__ v8f splat8(float x) {
  v8f c;
  #pragma unroll
  for (int i = 0; i < 8; ++i) c[i] = x;
  return c;
}

__device__ __forceinline__ v8f relu8(v8f c) {
  #pragma unroll
  for (int i = 0; i < 8; ++i) c[i] = (c[i] > 0.0f) ? c[i] : 0.0f;
  return c;
}

// ---------------------------------------------------------------------------
// Weight prepack: W[layer][K][N] (f32) -> WMMA B fragments (f16), with the
// bias injected as virtual weight row `biasRow` (pair with feature==1.0).
// Fragment f = (layer*KT + kt)*NT + nt ; lane l holds n = nt*16 + (l&15),
// element i <- K = kt*32 + (l>>4)*8 + i (i<8), kt*32 + 16 + (l>>4)*8 + (i-8) (i>=8). 1024 B per fragment.
// ---------------------------------------------------------------------------
__global__ void pack_b_kernel(const float* __restrict__ W,
                              const float* __restrict__ B,
                              _Float16* __restrict__ dst,
                              int K, int N, int KT, int NT, int biasRow) {
  int f = blockIdx.x;
  int nt = f % NT;
  int kt = (f / NT) % KT;
  int layer = f / (NT * KT);
  const float* Wl = W + (long)layer * K * N;
  int lane = threadIdx.x;
  int n = nt * 16 + (lane & 15);
  V16U v;
  _Float16 tmp[16];
  #pragma unroll
  for (int i = 0; i < 16; ++i) {
    int k = kt * 32 + (i < 8 ? ((lane >> 4) * 8) + i : 16 + ((lane >> 4) * 8) + (i - 8));
    float val = 0.0f;
    if (n < N) {
      if (k < K)            val = Wl[k * N + n];
      else if (k == biasRow) val = B[(long)layer * N + n];
    }
    tmp[i] = (_Float16)val;
  }
  #pragma unroll
  for (int i = 0; i < 16; ++i) ((_Float16*)&v)[i] = tmp[i];
  *((v16h*)(dst + (long)f * 512) + lane) = v.h;
}

__global__ void zero_f32_kernel(float* __restrict__ p, int n) {
  int i = blockIdx.x * blockDim.x + threadIdx.x;
  if (i < n) p[i] = 0.0f;
}

// ---------------------------------------------------------------------------
// Encoder node MLP: h = relu(x@W1+b1)@W2+b2   (18 -> 64 -> 18), b1 folded.
// ---------------------------------------------------------------------------
__global__ __launch_bounds__(256) void encoder_node_kernel(
    const float* __restrict__ x, float* __restrict__ h,
    const _Float16* __restrict__ w1p,
    const _Float16* __restrict__ w2p, const float* __restrict__ b2, int nNodes) {
  __shared__ __align__(16) _Float16 sF[WV][16][32];
  __shared__ __align__(16) _Float16 sH[WV][16][64];
  int lane = threadIdx.x & 31, wave = threadIdx.x >> 5;
  int base = (blockIdx.x * WV + wave) * 16;
  if (base >= nNodes) return;

  int n = lane & 15, kh = lane >> 4;
  {
    int nid = base + n; if (nid >= nNodes) nid = nNodes - 1;
    _Float16* row = sF[wave][n];
    if (kh == 0) {                     // lanes 0-15: load the 18 features
      const float* xr = x + (long)nid * 18;
      #pragma unroll
      for (int k = 0; k < 9; ++k) {
        float2 a = *(const float2*)(xr + 2 * k);
        row[2 * k] = (_Float16)a.x; row[2 * k + 1] = (_Float16)a.y;
      }
    } else {                           // lanes 16-31: bias row + padding
      row[18] = (_Float16)1.0f;
      #pragma unroll
      for (int k = 19; k < 32; ++k) row[k] = (_Float16)0.0f;
    }
  }
  v16h a = lds_a_frag(&sF[wave][0][0], 32, 0, lane);
  _Float16* hid = &sH[wave][0][0];
  #pragma unroll
  for (int t = 0; t < 4; ++t) {               // hidden = 64 -> 4 N-tiles
    v8f c = {};
    c = wmma_f16(a, ld_b_frag(w1p + t * 512, lane), c);
    c = relu8(c);
    #pragma unroll
    for (int r = 0; r < 8; ++r)
      hid[(r + 8 * kh) * 64 + t * 16 + n] = (_Float16)c[r];
  }
  v16h a0 = lds_a_frag(hid, 64, 0, lane);
  v16h a1 = lds_a_frag(hid, 64, 32, lane);
  v8f d0 = splat8(b2[n]);
  d0 = wmma_f16(a0, ld_b_frag(w2p + 0 * 512, lane), d0);  // kt0,nt0
  d0 = wmma_f16(a1, ld_b_frag(w2p + 2 * 512, lane), d0);  // kt1,nt0
  v8f d1 = splat8(n < 2 ? b2[16 + n] : 0.0f);
  d1 = wmma_f16(a0, ld_b_frag(w2p + 1 * 512, lane), d1);  // kt0,nt1
  d1 = wmma_f16(a1, ld_b_frag(w2p + 3 * 512, lane), d1);  // kt1,nt1
  #pragma unroll
  for (int r = 0; r < 8; ++r) {
    int nid = base + r + 8 * kh;
    if (nid < nNodes) {
      h[(long)nid * 18 + n] = d0[r];
      if (n < 2) h[(long)nid * 18 + 16 + n] = d1[r];
    }
  }
}

// ---------------------------------------------------------------------------
// Encoder edge MLP: e = relu(ea@W1+b1)@W2+b2   (3 -> 64 -> 3), b1 folded.
// ---------------------------------------------------------------------------
__global__ __launch_bounds__(256) void encoder_edge_kernel(
    const float* __restrict__ ea, float* __restrict__ e,
    const _Float16* __restrict__ w1p,
    const _Float16* __restrict__ w2p, const float* __restrict__ b2, int nEdges) {
  __shared__ __align__(16) _Float16 sF[WV][16][32];
  __shared__ __align__(16) _Float16 sH[WV][16][64];
  int lane = threadIdx.x & 31, wave = threadIdx.x >> 5;
  int base = (blockIdx.x * WV + wave) * 16;
  if (base >= nEdges) return;

  int n = lane & 15, kh = lane >> 4;
  {
    int eid = base + n; if (eid >= nEdges) eid = nEdges - 1;
    _Float16* row = sF[wave][n];
    if (kh == 0) {
      const float* er = ea + (long)eid * 3;
      row[0] = (_Float16)er[0]; row[1] = (_Float16)er[1]; row[2] = (_Float16)er[2];
      row[3] = (_Float16)1.0f;                 // bias feature
      #pragma unroll
      for (int k = 4; k < 16; ++k) row[k] = (_Float16)0.0f;
    } else {
      #pragma unroll
      for (int k = 16; k < 32; ++k) row[k] = (_Float16)0.0f;
    }
  }
  v16h a = lds_a_frag(&sF[wave][0][0], 32, 0, lane);
  _Float16* hid = &sH[wave][0][0];
  #pragma unroll
  for (int t = 0; t < 4; ++t) {
    v8f c = {};
    c = wmma_f16(a, ld_b_frag(w1p + t * 512, lane), c);
    c = relu8(c);
    #pragma unroll
    for (int r = 0; r < 8; ++r)
      hid[(r + 8 * kh) * 64 + t * 16 + n] = (_Float16)c[r];
  }
  v16h a0 = lds_a_frag(hid, 64, 0, lane);
  v16h a1 = lds_a_frag(hid, 64, 32, lane);
  v8f d = splat8(n < 3 ? b2[n] : 0.0f);
  d = wmma_f16(a0, ld_b_frag(w2p + 0 * 512, lane), d);
  d = wmma_f16(a1, ld_b_frag(w2p + 1 * 512, lane), d);
  if (n < 3) {
    #pragma unroll
    for (int r = 0; r < 8; ++r) {
      int eid = base + r + 8 * kh;
      if (eid < nEdges) e[(long)eid * 3 + n] = d[r];
    }
  }
}

// ---------------------------------------------------------------------------
// Processor edge pass: msg = relu([h[tgt],h[src],e]@W1+b1)@W2+b2 (39->32->3);
// e += msg ; aggr[tgt] += msg (atomic). b1 folded into K row 39.
// Gather split across the full wave: lane m loads x_i, lane m+16 loads x_j+e.
// ---------------------------------------------------------------------------
__global__ __launch_bounds__(256) void edge_layer_kernel(
    const float* __restrict__ h, float* __restrict__ e, float* __restrict__ aggr,
    const int* __restrict__ srcIdx, const int* __restrict__ tgtIdx,
    const _Float16* __restrict__ w1p,
    const _Float16* __restrict__ w2p, const float* __restrict__ b2, int nEdges) {
  __shared__ __align__(16) _Float16 sF[WV][16][64];
  __shared__ __align__(16) _Float16 sH[WV][16][32];
  __shared__ int   sTgt[WV][16];
  __shared__ float sEold[WV][16][3];
  int lane = threadIdx.x & 31, wave = threadIdx.x >> 5;
  int base = (blockIdx.x * WV + wave) * 16;
  if (base >= nEdges) return;

  int n = lane & 15, kh = lane >> 4;
  {
    int eid = base + n;
    int ec = (eid < nEdges) ? eid : (nEdges - 1);
    // Prefetch streamed arrays a few tiles ahead (global_prefetch_b8).
    __builtin_prefetch(srcIdx + ec + 2048, 0, 1);
    __builtin_prefetch(e + (long)ec * 3 + 6144, 0, 1);
    _Float16* row = sF[wave][n];
    if (kh == 0) {                     // x_i = h[tgt]
      int t = tgtIdx[ec];
      sTgt[wave][n] = t;
      const float* hi = h + (long)t * 18;
      #pragma unroll
      for (int k = 0; k < 9; ++k) {
        float2 a = *(const float2*)(hi + 2 * k);
        row[2 * k] = (_Float16)a.x; row[2 * k + 1] = (_Float16)a.y;
      }
    } else {                           // x_j = h[src], edge feats, bias, pad
      int s = srcIdx[ec];
      const float* hj = h + (long)s * 18;
      #pragma unroll
      for (int k = 0; k < 9; ++k) {
        float2 a = *(const float2*)(hj + 2 * k);
        row[18 + 2 * k] = (_Float16)a.x; row[18 + 2 * k + 1] = (_Float16)a.y;
      }
      const float* ep = e + (long)ec * 3;
      float e0 = ep[0], e1 = ep[1], e2 = ep[2];
      sEold[wave][n][0] = e0; sEold[wave][n][1] = e1; sEold[wave][n][2] = e2;
      row[36] = (_Float16)e0; row[37] = (_Float16)e1; row[38] = (_Float16)e2;
      row[39] = (_Float16)1.0f;        // bias feature
      #pragma unroll
      for (int k = 40; k < 64; ++k) row[k] = (_Float16)0.0f;
    }
  }
  const _Float16* fb = &sF[wave][0][0];
  v16h a0 = lds_a_frag(fb, 64, 0, lane);
  v16h a1 = lds_a_frag(fb, 64, 32, lane);
  v8f c0 = {}, c1 = {};
  c0 = wmma_f16(a0, ld_b_frag(w1p + 0 * 512, lane), c0);  // kt0,nt0
  c1 = wmma_f16(a0, ld_b_frag(w1p + 1 * 512, lane), c1);  // kt0,nt1
  c0 = wmma_f16(a1, ld_b_frag(w1p + 2 * 512, lane), c0);  // kt1,nt0
  c1 = wmma_f16(a1, ld_b_frag(w1p + 3 * 512, lane), c1);  // kt1,nt1
  c0 = relu8(c0); c1 = relu8(c1);
  _Float16* hid = &sH[wave][0][0];
  #pragma unroll
  for (int r = 0; r < 8; ++r) {
    hid[(r + 8 * kh) * 32 + n] = (_Float16)c0[r];
    hid[(r + 8 * kh) * 32 + 16 + n] = (_Float16)c1[r];
  }
  v16h ah = lds_a_frag(hid, 32, 0, lane);
  v8f m = splat8(n < 3 ? b2[n] : 0.0f);
  m = wmma_f16(ah, ld_b_frag(w2p, lane), m);
  if (n < 3) {
    #pragma unroll
    for (int r = 0; r < 8; ++r) {
      int mrow = r + 8 * kh;
      int eid = base + mrow;
      if (eid < nEdges) {
        float v = m[r];
        e[(long)eid * 3 + n] = sEold[wave][mrow][n] + v;      // e += msg
        atomicAdd(&aggr[(long)sTgt[wave][mrow] * 3 + n], v);  // segment_sum
      }
    }
  }
}

// ---------------------------------------------------------------------------
// Processor node pass: h += relu([h,aggr]@W1+b1)@W2+b2  (21 -> 32 -> 18)
// b1 folded into K row 21.
// ---------------------------------------------------------------------------
__global__ __launch_bounds__(256) void node_layer_kernel(
    float* __restrict__ h, const float* __restrict__ aggr,
    const _Float16* __restrict__ w1p,
    const _Float16* __restrict__ w2p, const float* __restrict__ b2, int nNodes) {
  __shared__ __align__(16) _Float16 sF[WV][16][32];
  __shared__ __align__(16) _Float16 sH[WV][16][32];
  int lane = threadIdx.x & 31, wave = threadIdx.x >> 5;
  int base = (blockIdx.x * WV + wave) * 16;
  if (base >= nNodes) return;

  int n = lane & 15, kh = lane >> 4;
  {
    int nid = base + n; if (nid >= nNodes) nid = nNodes - 1;
    _Float16* row = sF[wave][n];
    if (kh == 0) {                     // node features
      const float* hr = h + (long)nid * 18;
      #pragma unroll
      for (int k = 0; k < 9; ++k) {
        float2 a = *(const float2*)(hr + 2 * k);
        row[2 * k] = (_Float16)a.x; row[2 * k + 1] = (_Float16)a.y;
      }
    } else {                           // aggregated messages, bias, pad
      const float* ar = aggr + (long)nid * 3;
      row[18] = (_Float16)ar[0]; row[19] = (_Float16)ar[1]; row[20] = (_Float16)ar[2];
      row[21] = (_Float16)1.0f;        // bias feature
      #pragma unroll
      for (int k = 22; k < 32; ++k) row[k] = (_Float16)0.0f;
    }
  }
  v16h a = lds_a_frag(&sF[wave][0][0], 32, 0, lane);
  v8f c0 = {}, c1 = {};
  c0 = wmma_f16(a, ld_b_frag(w1p + 0 * 512, lane), c0);
  c1 = wmma_f16(a, ld_b_frag(w1p + 1 * 512, lane), c1);
  c0 = relu8(c0); c1 = relu8(c1);
  _Float16* hid = &sH[wave][0][0];
  #pragma unroll
  for (int r = 0; r < 8; ++r) {
    hid[(r + 8 * kh) * 32 + n] = (_Float16)c0[r];
    hid[(r + 8 * kh) * 32 + 16 + n] = (_Float16)c1[r];
  }
  v16h ah = lds_a_frag(hid, 32, 0, lane);
  v8f d0 = splat8(b2[n]);
  v8f d1 = splat8(n < 2 ? b2[16 + n] : 0.0f);
  d0 = wmma_f16(ah, ld_b_frag(w2p + 0 * 512, lane), d0);
  d1 = wmma_f16(ah, ld_b_frag(w2p + 1 * 512, lane), d1);
  #pragma unroll
  for (int r = 0; r < 8; ++r) {
    int nid = base + r + 8 * kh;
    if (nid < nNodes) {
      h[(long)nid * 18 + n] += d0[r];                 // residual
      if (n < 2) h[(long)nid * 18 + 16 + n] += d1[r];
    }
  }
}

// ---------------------------------------------------------------------------
// Decoder: acc = relu(h@W1+b1)@W2+b2   (18 -> 16 -> 3), both biases folded
// (b1 -> K row 18 of W1; b2 -> K row 16 of W2, hidden col 16 := 1.0).
// ---------------------------------------------------------------------------
__global__ __launch_bounds__(256) void decoder_kernel(
    const float* __restrict__ h, float* __restrict__ out,
    const _Float16* __restrict__ w1p,
    const _Float16* __restrict__ w2p, int nNodes) {
  __shared__ __align__(16) _Float16 sF[WV][16][32];
  __shared__ __align__(16) _Float16 sH[WV][16][32];
  int lane = threadIdx.x & 31, wave = threadIdx.x >> 5;
  int base = (blockIdx.x * WV + wave) * 16;
  if (base >= nNodes) return;

  int n = lane & 15, kh = lane >> 4;
  {
    int nid = base + n; if (nid >= nNodes) nid = nNodes - 1;
    _Float16* row = sF[wave][n];
    if (kh == 0) {
      const float* hr = h + (long)nid * 18;
      #pragma unroll
      for (int k = 0; k < 9; ++k) {
        float2 a = *(const float2*)(hr + 2 * k);
        row[2 * k] = (_Float16)a.x; row[2 * k + 1] = (_Float16)a.y;
      }
    } else {
      row[18] = (_Float16)1.0f;        // bias feature
      #pragma unroll
      for (int k = 19; k < 32; ++k) row[k] = (_Float16)0.0f;
    }
  }
  v16h a = lds_a_frag(&sF[wave][0][0], 32, 0, lane);
  v8f c = {};
  c = wmma_f16(a, ld_b_frag(w1p, lane), c);
  c = relu8(c);
  _Float16* hid = &sH[wave][0][0];
  #pragma unroll
  for (int r = 0; r < 8; ++r) {
    hid[(r + 8 * kh) * 32 + n] = (_Float16)c[r];                  // hidden = 16
    hid[(r + 8 * kh) * 32 + 16 + n] = (_Float16)(n == 0 ? 1.0f : 0.0f); // bias col
  }
  v16h ah = lds_a_frag(hid, 32, 0, lane);
  v8f d = {};
  d = wmma_f16(ah, ld_b_frag(w2p, lane), d);
  if (n < 3) {
    #pragma unroll
    for (int r = 0; r < 8; ++r) {
      int nid = base + r + 8 * kh;
      if (nid < nNodes) out[(long)nid * 3 + n] = d[r];
    }
  }
}

// ---------------------------------------------------------------------------
// Host orchestration
// d_in order (setup_inputs dict, params in insertion order):
//  0:x 1:edge_attr 2..5:en_w1,en_b1,en_w2,en_b2 6..9:ee_* 10..13:pe_w1,pe_b1,
//  pe_w2,pe_b2 14..17:pn_* 18..21:d_* 22:edge_index[2,E]
// ---------------------------------------------------------------------------
extern "C" void kernel_launch(void* const* d_in, const int* in_sizes, int n_in,
                              void* d_out, int out_size, void* d_ws, size_t ws_size,
                              hipStream_t stream) {
  const float* x     = (const float*)d_in[0];
  const float* ea    = (const float*)d_in[1];
  const float* en_w1 = (const float*)d_in[2];
  const float* en_b1 = (const float*)d_in[3];
  const float* en_w2 = (const float*)d_in[4];
  const float* en_b2 = (const float*)d_in[5];
  const float* ee_w1 = (const float*)d_in[6];
  const float* ee_b1 = (const float*)d_in[7];
  const float* ee_w2 = (const float*)d_in[8];
  const float* ee_b2 = (const float*)d_in[9];
  const float* pe_w1 = (const float*)d_in[10];
  const float* pe_b1 = (const float*)d_in[11];
  const float* pe_w2 = (const float*)d_in[12];
  const float* pe_b2 = (const float*)d_in[13];
  const float* pn_w1 = (const float*)d_in[14];
  const float* pn_b1 = (const float*)d_in[15];
  const float* pn_w2 = (const float*)d_in[16];
  const float* pn_b2 = (const float*)d_in[17];
  const float* d_w1  = (const float*)d_in[18];
  const float* d_b1  = (const float*)d_in[19];
  const float* d_w2  = (const float*)d_in[20];
  const float* d_b2  = (const float*)d_in[21];
  const int*   eidx  = (const int*)d_in[22];

  const int nNodes = in_sizes[0] / 18;
  const int nEdges = in_sizes[1] / 3;
  const int* srcIdx = eidx;            // edge_index[0]
  const int* tgtIdx = eidx + nEdges;   // edge_index[1]

  // Workspace layout
  char* ws = (char*)d_ws;
  float* h    = (float*)(ws);
  float* e    = (float*)(ws + (size_t)nNodes * 18 * 4);
  float* aggr = (float*)(ws + (size_t)nNodes * 18 * 4 + (size_t)nEdges * 3 * 4);
  _Float16* pk = (_Float16*)(ws + (size_t)nNodes * 18 * 4 +
                             (size_t)nEdges * 3 * 4 + (size_t)nNodes * 3 * 4);
  _Float16* pEnW1 = pk;                  // 1*4  frags
  _Float16* pEnW2 = pEnW1 + 4  * 512;    // 2*2
  _Float16* pEeW1 = pEnW2 + 4  * 512;    // 1*4
  _Float16* pEeW2 = pEeW1 + 4  * 512;    // 2*1
  _Float16* pPeW1 = pEeW2 + 2  * 512;    // 10 * 2*2
  _Float16* pPeW2 = pPeW1 + 40 * 512;    // 10 * 1*1
  _Float16* pPnW1 = pPeW2 + 10 * 512;    // 10 * 1*2
  _Float16* pPnW2 = pPnW1 + 20 * 512;    // 10 * 1*2
  _Float16* pDW1  = pPnW2 + 20 * 512;    // 1
  _Float16* pDW2  = pDW1  + 1  * 512;    // 1

  // Prepack all weights into WMMA-B fragment order (f16); fold first-layer
  // biases into virtual K row (feature == 1.0). biasRow = -1 disables.
  pack_b_kernel<<<4,  32, 0, stream>>>(en_w1, en_b1, pEnW1, 18, 64, 1, 4, 18);
  pack_b_kernel<<<4,  32, 0, stream>>>(en_w2, nullptr, pEnW2, 64, 18, 2, 2, -1);
  pack_b_kernel<<<4,  32, 0, stream>>>(ee_w1, ee_b1, pEeW1,  3, 64, 1, 4,  3);
  pack_b_kernel<<<2,  32, 0, stream>>>(ee_w2, nullptr, pEeW2, 64,  3, 2, 1, -1);
  pack_b_kernel<<<40, 32, 0, stream>>>(pe_w1, pe_b1, pPeW1, 39, 32, 2, 2, 39);
  pack_b_kernel<<<10, 32, 0, stream>>>(pe_w2, nullptr, pPeW2, 32,  3, 1, 1, -1);
  pack_b_kernel<<<20, 32, 0, stream>>>(pn_w1, pn_b1, pPnW1, 21, 32, 1, 2, 21);
  pack_b_kernel<<<20, 32, 0, stream>>>(pn_w2, nullptr, pPnW2, 32, 18, 1, 2, -1);
  pack_b_kernel<<<1,  32, 0, stream>>>(d_w1,  d_b1,  pDW1,  18, 16, 1, 1, 18);
  pack_b_kernel<<<1,  32, 0, stream>>>(d_w2,  d_b2,  pDW2,  16,  3, 1, 1, 16);

  const int nodeTiles  = (nNodes + 15) / 16;
  const int nodeBlocks = (nodeTiles + WV - 1) / WV;
  const int edgeTiles  = (nEdges + 15) / 16;
  const int edgeBlocks = (edgeTiles + WV - 1) / WV;

  encoder_node_kernel<<<nodeBlocks, 256, 0, stream>>>(x, h, pEnW1,
                                                      pEnW2, en_b2, nNodes);
  encoder_edge_kernel<<<edgeBlocks, 256, 0, stream>>>(ea, e, pEeW1,
                                                      pEeW2, ee_b2, nEdges);

  for (int i = 0; i < 10; ++i) {
    zero_f32_kernel<<<(nNodes * 3 + 1023) / 1024, 1024, 0, stream>>>(aggr,
                                                                     nNodes * 3);
    edge_layer_kernel<<<edgeBlocks, 256, 0, stream>>>(
        h, e, aggr, srcIdx, tgtIdx,
        pPeW1 + (size_t)i * 4 * 512,
        pPeW2 + (size_t)i * 512, pe_b2 + i * 3, nEdges);
    node_layer_kernel<<<nodeBlocks, 256, 0, stream>>>(
        h, aggr,
        pPnW1 + (size_t)i * 2 * 512,
        pPnW2 + (size_t)i * 2 * 512, pn_b2 + i * 18, nNodes);
  }

  decoder_kernel<<<nodeBlocks, 256, 0, stream>>>(h, (float*)d_out, pDW1,
                                                 pDW2, nNodes);
}
